// Block_3504693313975
// MI455X (gfx1250) — compile-verified, matrix operand order checked by reading, not run on hardware
//
#include <hip/hip_runtime.h>

// Block: B=16, T=1024, D=448, H=8, HS=56 (padded to 64), Dff=1792
#define B_    16
#define T_    1024
#define D_    448
#define H_    8
#define HS_   56
#define HSP_  64
#define DH_   512            // H_*HSP_
#define DFF_  1792
#define NTOK_ 16384          // B_*T_

typedef __attribute__((ext_vector_type(16))) __bf16        v16bf;
typedef __attribute__((ext_vector_type(8)))  float         v8f;
typedef __attribute__((ext_vector_type(4)))  unsigned int  u32x4;

union Frag { v16bf v; u32x4 u[2]; };   // 16 bf16 = 8 VGPRs per lane
union Pack8 { unsigned short s[8]; u32x4 u; };

__device__ __forceinline__ unsigned short f2bf(float f) {
  unsigned int x = __float_as_uint(f);
  x += 0x7fffu + ((x >> 16) & 1u);     // round-to-nearest-even
  return (unsigned short)(x >> 16);
}

// ---------------- DPP 16-lane row reductions (pure VALU, no LDS) ----------
// quad_perm(1,0,3,2)=0xB1 (xor1), quad_perm(2,3,0,1)=0x4E (xor2),
// row_ror:4=0x124, row_ror:8=0x128 fold quad-uniform partials across the row.
template <int CTRL>
__device__ __forceinline__ float dppmovf(float x) {
  return __int_as_float(__builtin_amdgcn_update_dpp(
      __float_as_int(x), __float_as_int(x), CTRL, 0xF, 0xF, true));
}
__device__ __forceinline__ float redmax16(float x) {
  x = fmaxf(x, dppmovf<0xB1>(x));
  x = fmaxf(x, dppmovf<0x4E>(x));
  x = fmaxf(x, dppmovf<0x124>(x));
  x = fmaxf(x, dppmovf<0x128>(x));
  return x;
}
__device__ __forceinline__ float redsum16(float x) {
  x += dppmovf<0xB1>(x);
  x += dppmovf<0x4E>(x);
  x += dppmovf<0x124>(x);
  x += dppmovf<0x128>(x);
  return x;
}
__device__ __forceinline__ float redsum32(float x) {
  x = redsum16(x);
  x += __shfl_xor(x, 16, 32);          // single cross-half hop
  return x;
}

// ---------------- elementwise prep kernels ----------------

__global__ void cvt_bf16_kernel(const float* __restrict__ in,
                                unsigned short* __restrict__ out, int n) {
  int i = blockIdx.x * blockDim.x + threadIdx.x;
  if (i < n) out[i] = f2bf(in[i]);
}

// wq/wk/wv [H][56][448] fp32 -> wpad [3][H][64][448] bf16, rows 56..63 zeroed
__global__ void pack_qkv_kernel(const float* __restrict__ wq,
                                const float* __restrict__ wk,
                                const float* __restrict__ wv,
                                unsigned short* __restrict__ wpad) {
  int i = blockIdx.x * blockDim.x + threadIdx.x;      // 3*512*448 total
  const int per = DH_ * D_;
  int which = i / per;
  int rem   = i - which * per;
  int n = rem / D_;            // 0..511
  int c = rem - n * D_;        // 0..447
  int h = n >> 6;
  int r = n & 63;
  const float* src = (which == 0) ? wq : (which == 1) ? wk : wv;
  wpad[i] = (r < HS_) ? f2bf(src[(h * HS_ + r) * D_ + c]) : (unsigned short)0;
}

// one wave per row: D=448 = 14*32, DPP + one shuffle reduction
__global__ void ln_bf16_kernel(const float* __restrict__ x,
                               const float* __restrict__ gamma,
                               const float* __restrict__ beta,
                               unsigned short* __restrict__ out) {
  const int wave = threadIdx.x >> 5;
  const int lane = threadIdx.x & 31;
  const int row  = blockIdx.x * 8 + wave;
  const float* xr = x + (size_t)row * D_;
  float v[14];
  float s = 0.f;
#pragma unroll
  for (int j = 0; j < 14; ++j) { v[j] = xr[lane + 32 * j]; s += v[j]; }
  s = redsum32(s);
  float mu = s * (1.0f / D_);
  float q = 0.f;
#pragma unroll
  for (int j = 0; j < 14; ++j) { float d = v[j] - mu; q += d * d; }
  q = redsum32(q);
  float rstd = rsqrtf(q * (1.0f / D_) + 1e-5f);
  unsigned short* orow = out + (size_t)row * D_;
#pragma unroll
  for (int j = 0; j < 14; ++j) {
    int c = lane + 32 * j;
    orow[c] = f2bf((v[j] - mu) * rstd * gamma[c] + beta[c]);
  }
}

// ---------------- WMMA GEMM: C[M,N] = A[M,K] * Bw[N,K]^T (+epilogue) --------
// MODE 0: bf16 out (no bias)                -> Q/K projections
// MODE 1: bf16 relu(out + bias)             -> FFN1
// MODE 2: f32  out + bias + res             -> proj / FFN2 (+residual)
// MODE 3: bf16 out, transposed V layout     -> V projection into [B,H,64,T]
template <int MODE>
__global__ void gemm_bf16_wmma(const unsigned short* __restrict__ A,
                               const unsigned short* __restrict__ Bw,
                               const float* __restrict__ bias,
                               const float* __restrict__ res,
                               void* __restrict__ outp, int N, int K) {
  __shared__ __align__(16) unsigned short sA[128][32];
  __shared__ __align__(16) unsigned short sB[64][32];

  const int blockM = blockIdx.y * 128;
  const int blockN = blockIdx.x * 64;
  const int tid  = threadIdx.x;
  const int wave = tid >> 5;
  const int lane = tid & 31;
  const int lm = lane & 15;
  const int hi = lane >> 4;
  const int waveM = (wave >> 1) * 32;   // 4 wave-rows
  const int waveN = (wave & 1) * 32;    // 2 wave-cols

  const v8f vz = {0.f, 0.f, 0.f, 0.f, 0.f, 0.f, 0.f, 0.f};
  v8f c[2][2];
#pragma unroll
  for (int i = 0; i < 2; ++i)
#pragma unroll
    for (int j = 0; j < 2; ++j) c[i][j] = vz;

  const int arow = tid >> 1, acol = (tid & 1) * 16;   // 128x32 A tile
  const int brow = tid >> 2, bcol = (tid & 3) * 8;    // 64x32  B tile
  const unsigned short* ga = A  + (size_t)(blockM + arow) * K + acol;
  const unsigned short* gb = Bw + (size_t)(blockN + brow) * K + bcol;

  for (int k0 = 0; k0 < K; k0 += 32) {
    *(u32x4*)&sA[arow][acol]     = *(const u32x4*)(ga + k0);
    *(u32x4*)&sA[arow][acol + 8] = *(const u32x4*)(ga + k0 + 8);
    *(u32x4*)&sB[brow][bcol]     = *(const u32x4*)(gb + k0);
    if (k0 + 32 < K) {                       // prefetch next K tile
      __builtin_prefetch(ga + k0 + 32, 0, 0);
      __builtin_prefetch(gb + k0 + 32, 0, 0);
    }
    __syncthreads();

    Frag a[2], b[2];
#pragma unroll
    for (int i = 0; i < 2; ++i) {
      const unsigned short* p = &sA[waveM + i * 16 + lm][0];
      a[i].u[0] = *(const u32x4*)(p + hi * 8);
      a[i].u[1] = *(const u32x4*)(p + 16 + hi * 8);
    }
#pragma unroll
    for (int j = 0; j < 2; ++j) {
      const unsigned short* p = &sB[waveN + j * 16 + lm][0];
      b[j].u[0] = *(const u32x4*)(p + hi * 8);
      b[j].u[1] = *(const u32x4*)(p + 16 + hi * 8);
    }
#pragma unroll
    for (int i = 0; i < 2; ++i)
#pragma unroll
      for (int j = 0; j < 2; ++j)
        c[i][j] = __builtin_amdgcn_wmma_f32_16x16x32_bf16(
            false, a[i].v, false, b[j].v, (short)0, c[i][j], false, false);
    __syncthreads();
  }

#pragma unroll
  for (int i = 0; i < 2; ++i) {
#pragma unroll
    for (int j = 0; j < 2; ++j) {
      const int col = blockN + waveN + j * 16 + lm;
      if (MODE == 3) {
        // lane holds 8 consecutive rows (t values) of one (h,e) column:
        // pack and store 16B directly into vT[B,H,64,T]
        const int row0 = blockM + waveM + i * 16 + 8 * hi;   // multiple of 8
        Pack8 pk;
#pragma unroll
        for (int r = 0; r < 8; ++r) pk.s[r] = f2bf(c[i][j][r]);
        const size_t vidx =
            ((size_t)(row0 >> 10) * DH_ + col) * T_ + (row0 & (T_ - 1));
        *(u32x4*)((unsigned short*)outp + vidx) = pk.u;
      } else {
        const float bs = (MODE == 0) ? 0.0f : bias[col];
#pragma unroll
        for (int r = 0; r < 8; ++r) {
          const int row = blockM + waveM + i * 16 + r + 8 * hi;
          const size_t idx = (size_t)row * N + col;
          float val = c[i][j][r] + bs;
          if (MODE == 0) {
            ((unsigned short*)outp)[idx] = f2bf(val);
          } else if (MODE == 1) {
            ((unsigned short*)outp)[idx] = f2bf(val > 0.f ? val : 0.f);
          } else {
            ((float*)outp)[idx] = val + res[idx];
          }
        }
      }
    }
  }
}

// ---------------- flash attention: 1 wave = 16 Q rows ----------------------
// q,k: [B*T, 512] bf16 (b,t,h,e padded 64); vT: [B,H,64,T] bf16
// out: attn concat [B*T, 448] bf16

// One 32-key block: S = Q*K^T, online softmax, O += P*V.
// MASK=true only for the diagonal block -> mask logic stays out of the
// steady-state loop entirely.
template <bool MASK>
__device__ __forceinline__ void attn_step(
    int key0, int qt0, int lm, int hi, int b, int h, int bh,
    const unsigned short* __restrict__ k,
    const unsigned short* __restrict__ vT,
    unsigned short* __restrict__ pw,                 // this wave's 16x32 P tile
    const Frag* __restrict__ qa, v8f* __restrict__ o,
    float* __restrict__ mi, float* __restrict__ li) {
  const float scale = 0.0472455591f;                 // 1/sqrt(448)
  const v8f vz = {0.f, 0.f, 0.f, 0.f, 0.f, 0.f, 0.f, 0.f};

  // S = Q * K^T : two 16x16 C tiles
  v8f s[2];
#pragma unroll
  for (int jj = 0; jj < 2; ++jj) {
    Frag kf0, kf1;
    const unsigned short* kp =
        k + (size_t)(b * T_ + key0 + jj * 16 + lm) * DH_ + h * HSP_;
    kf0.u[0] = *(const u32x4*)(kp + hi * 8);
    kf0.u[1] = *(const u32x4*)(kp + 16 + hi * 8);
    kf1.u[0] = *(const u32x4*)(kp + 32 + hi * 8);
    kf1.u[1] = *(const u32x4*)(kp + 48 + hi * 8);
    v8f t0 = __builtin_amdgcn_wmma_f32_16x16x32_bf16(
        false, qa[0].v, false, kf0.v, (short)0, vz, false, false);
    s[jj] = __builtin_amdgcn_wmma_f32_16x16x32_bf16(
        false, qa[1].v, false, kf1.v, (short)0, t0, false, false);
  }

#pragma unroll
  for (int r = 0; r < 8; ++r) {
    float s0 = s[0][r] * scale;
    float s1 = s[1][r] * scale;
    if (MASK) {
      const int row = qt0 + r + 8 * hi;
      if (key0 + lm > row)      s0 = -1e30f;
      if (key0 + 16 + lm > row) s1 = -1e30f;
    }
    float rm = redmax16(fmaxf(s0, s1));              // DPP, no LDS
    float mnew  = fmaxf(mi[r], rm);
    float alpha = __expf(mi[r] - mnew);
    float p0 = __expf(s0 - mnew);
    float p1 = __expf(s1 - mnew);
    float rs = redsum16(p0 + p1);                    // DPP, no LDS
    li[r] = li[r] * alpha + rs;
    mi[r] = mnew;
#pragma unroll
    for (int t = 0; t < 4; ++t) o[t][r] *= alpha;
    // write P to LDS right away (no spill buffers)
    pw[(r + 8 * hi) * 32 + lm]      = f2bf(p0);
    pw[(r + 8 * hi) * 32 + 16 + lm] = f2bf(p1);
  }
  // per-wave DS ordering; explicit wait keeps the reload after the stores
  asm volatile("s_wait_dscnt 0" ::: "memory");
  Frag pa;
  pa.u[0] = *(const u32x4*)(pw + lm * 32 + hi * 8);
  pa.u[1] = *(const u32x4*)(pw + lm * 32 + 16 + hi * 8);

  // O += P * V  (V transposed: contiguous over keys)
#pragma unroll
  for (int t = 0; t < 4; ++t) {
    Frag vf;
    const unsigned short* vp =
        vT + ((size_t)bh * HSP_ + t * 16 + lm) * T_ + key0;
    vf.u[0] = *(const u32x4*)(vp + hi * 8);
    vf.u[1] = *(const u32x4*)(vp + 16 + hi * 8);
    o[t] = __builtin_amdgcn_wmma_f32_16x16x32_bf16(
        false, pa.v, false, vf.v, (short)0, o[t], false, false);
  }
}

__global__ void __launch_bounds__(128, 1)
flash_attn_kernel(const unsigned short* __restrict__ q,
                  const unsigned short* __restrict__ k,
                  const unsigned short* __restrict__ vT,
                  unsigned short* __restrict__ attn) {
  __shared__ __align__(16) unsigned short pbuf[4][16][32];   // per-wave P tile

  const int qblk = blockIdx.x & 15;       // T/64 = 16 blocks of 64 rows
  const int bh   = blockIdx.x >> 4;
  const int b = bh >> 3, h = bh & 7;
  const int wave = threadIdx.x >> 5;
  const int lane = threadIdx.x & 31;
  const int lm = lane & 15;
  const int hi = lane >> 4;
  const int qt0 = qblk * 64 + wave * 16;
  unsigned short* pw = &pbuf[wave][0][0];

  // Q A-fragments (16 rows x 64 head-dim = 2 K-chunks)
  Frag qa[2];
  {
    const unsigned short* qp = q + (size_t)(b * T_ + qt0 + lm) * DH_ + h * HSP_;
#pragma unroll
    for (int cch = 0; cch < 2; ++cch) {
      qa[cch].u[0] = *(const u32x4*)(qp + cch * 32 + hi * 8);
      qa[cch].u[1] = *(const u32x4*)(qp + cch * 32 + 16 + hi * 8);
    }
  }

  const v8f vz = {0.f, 0.f, 0.f, 0.f, 0.f, 0.f, 0.f, 0.f};
  v8f o[4];
#pragma unroll
  for (int t = 0; t < 4; ++t) o[t] = vz;
  float mi[8], li[8];
#pragma unroll
  for (int r = 0; r < 8; ++r) { mi[r] = -1e30f; li[r] = 0.f; }

  const int nkb = qt0 / 32 + 1;           // causal: 32-key blocks
  for (int kb = 0; kb < nkb - 1; ++kb)    // steady state: no masking
    attn_step<false>(kb * 32, qt0, lm, hi, b, h, bh, k, vT, pw, qa, o, mi, li);
  // diagonal block with causal mask
  attn_step<true>((nkb - 1) * 32, qt0, lm, hi, b, h, bh, k, vT, pw, qa, o, mi, li);

  // normalize and store concat heads (only e < 56)
#pragma unroll
  for (int t = 0; t < 4; ++t) {
    const int e = t * 16 + lm;
    if (e < HS_) {
#pragma unroll
      for (int r = 0; r < 8; ++r) {
        const int row = qt0 + r + 8 * hi;
        attn[(size_t)(b * T_ + row) * D_ + h * HS_ + e] =
            f2bf(o[t][r] / li[r]);
      }
    }
  }
}

// ---------------- launch ----------------

extern "C" void kernel_launch(void* const* d_in, const int* in_sizes, int n_in,
                              void* d_out, int out_size, void* d_ws,
                              size_t ws_size, hipStream_t stream) {
  (void)in_sizes; (void)n_in; (void)out_size; (void)ws_size;
  const float* x      = (const float*)d_in[0];
  const float* wq     = (const float*)d_in[1];
  const float* wk     = (const float*)d_in[2];
  const float* wv     = (const float*)d_in[3];
  const float* w_proj = (const float*)d_in[4];
  const float* b_proj = (const float*)d_in[5];
  const float* w1     = (const float*)d_in[6];
  const float* b1     = (const float*)d_in[7];
  const float* w2     = (const float*)d_in[8];
  const float* b2     = (const float*)d_in[9];
  const float* ln1_g  = (const float*)d_in[10];
  const float* ln1_b  = (const float*)d_in[11];
  const float* ln2_g  = (const float*)d_in[12];
  const float* ln2_b  = (const float*)d_in[13];
  float* out = (float*)d_out;

  char* ws = (char*)d_ws;
  size_t off = 0;
  auto alloc = [&](size_t bytes) -> void* {
    void* p = ws + off;
    off += (bytes + 255) & ~(size_t)255;
    return p;
  };
  unsigned short* h1     = (unsigned short*)alloc((size_t)NTOK_ * D_ * 2);
  unsigned short* wqkvp  = (unsigned short*)alloc((size_t)3 * DH_ * D_ * 2);
  unsigned short* wprojb = (unsigned short*)alloc((size_t)D_ * D_ * 2);
  unsigned short* w1b    = (unsigned short*)alloc((size_t)DFF_ * D_ * 2);
  unsigned short* w2b    = (unsigned short*)alloc((size_t)D_ * DFF_ * 2);
  unsigned short* qb     = (unsigned short*)alloc((size_t)NTOK_ * DH_ * 2);
  unsigned short* kb     = (unsigned short*)alloc((size_t)NTOK_ * DH_ * 2);
  unsigned short* vTb    = (unsigned short*)alloc((size_t)NTOK_ * DH_ * 2);
  unsigned short* attnb  = (unsigned short*)alloc((size_t)NTOK_ * D_ * 2);
  float*          x1     = (float*)alloc((size_t)NTOK_ * D_ * 4);
  unsigned short* h2     = (unsigned short*)alloc((size_t)NTOK_ * D_ * 2);
  unsigned short* ff1    = (unsigned short*)alloc((size_t)NTOK_ * DFF_ * 2);

  // weight prep (bf16 + head-dim padding)
  pack_qkv_kernel<<<(3 * DH_ * D_) / 256, 256, 0, stream>>>(wq, wk, wv, wqkvp);
  cvt_bf16_kernel<<<(D_ * D_ + 255) / 256, 256, 0, stream>>>(w_proj, wprojb, D_ * D_);
  cvt_bf16_kernel<<<(DFF_ * D_ + 255) / 256, 256, 0, stream>>>(w1, w1b, DFF_ * D_);
  cvt_bf16_kernel<<<(D_ * DFF_ + 255) / 256, 256, 0, stream>>>(w2, w2b, D_ * DFF_);

  // LN1 -> h1 (bf16)
  ln_bf16_kernel<<<NTOK_ / 8, 256, 0, stream>>>(x, ln1_g, ln1_b, h1);

  // QKV projections (V written directly in transposed [B,H,64,T] layout)
  dim3 gq(DH_ / 64, NTOK_ / 128);
  gemm_bf16_wmma<0><<<gq, 256, 0, stream>>>(h1, wqkvp,                nullptr, nullptr, qb,  DH_, D_);
  gemm_bf16_wmma<0><<<gq, 256, 0, stream>>>(h1, wqkvp + DH_ * D_,     nullptr, nullptr, kb,  DH_, D_);
  gemm_bf16_wmma<3><<<gq, 256, 0, stream>>>(h1, wqkvp + 2 * DH_ * D_, nullptr, nullptr, vTb, DH_, D_);

  // causal flash attention -> attnb (bf16, heads concatenated)
  flash_attn_kernel<<<B_ * H_ * (T_ / 64), 128, 0, stream>>>(qb, kb, vTb, attnb);

  // out proj + bias + residual(x) -> x1 (f32)
  dim3 gp(D_ / 64, NTOK_ / 128);
  gemm_bf16_wmma<2><<<gp, 256, 0, stream>>>(attnb, wprojb, b_proj, x, x1, D_, D_);

  // LN2 -> h2 (bf16)
  ln_bf16_kernel<<<NTOK_ / 8, 256, 0, stream>>>(x1, ln2_g, ln2_b, h2);

  // FFN1: relu(h2 @ w1^T + b1) -> ff1 (bf16)
  dim3 gf1(DFF_ / 64, NTOK_ / 128);
  gemm_bf16_wmma<1><<<gf1, 256, 0, stream>>>(h2, w1b, b1, nullptr, ff1, DFF_, D_);

  // FFN2: ff1 @ w2^T + b2 + residual(x1) -> out (f32)
  gemm_bf16_wmma<2><<<gp, 256, 0, stream>>>(ff1, w2b, b2, x1, out, D_, DFF_);
}
